// Decoder_32117765439553
// MI455X (gfx1250) — matrix operand order repaired and checked by reading, not compiled, not run on hardware
//
#include <hip/hip_runtime.h>
#include <hip/hip_bf16.h>

typedef __attribute__((ext_vector_type(16))) _Float16 v16h;
typedef __attribute__((ext_vector_type(8)))  _Float16 v8h;
typedef __attribute__((ext_vector_type(8)))  float    v8f;

#define VOCAB 32000
#define EMBD  256
#define HIDD  256
#define BB    32
#define SRCL  128
#define TT    64
#define KCLS  512   // 2*HID
#define MROW  2048  // B*T

__device__ __forceinline__ float sigmoidf_(float x) { return 1.0f / (1.0f + __expf(-x)); }

// ---------------------------------------------------------------------------
// Kernel 0: convert W_cls (fp32 [V, 512]) -> f16, 4 elems/thread
// ---------------------------------------------------------------------------
__global__ __launch_bounds__(256) void cvt_wcls_f16(const float* __restrict__ src,
                                                    _Float16* __restrict__ dst, int n) {
  int i4 = (blockIdx.x * 256 + threadIdx.x) * 4;
  if (i4 + 3 < n) {
    float4 v = *(const float4*)(src + i4);
    dst[i4 + 0] = (_Float16)v.x;
    dst[i4 + 1] = (_Float16)v.y;
    dst[i4 + 2] = (_Float16)v.z;
    dst[i4 + 3] = (_Float16)v.w;
  }
}

// ---------------------------------------------------------------------------
// Phase A: per-batch-element GRU + attention recurrence.
// grid = 32 blocks (one per b), 256 threads. Emits pred16 [B][T][512] f16.
// ---------------------------------------------------------------------------
__global__ __launch_bounds__(256) void decoder_recurrent(
    const float* __restrict__ enc,     // [B, SRC, H]
    const float* __restrict__ h_init,  // [B, H]
    const int*   __restrict__ tgt,     // [B, T]
    const float* __restrict__ emb,     // [VOCAB, E]
    const float* __restrict__ W_hm, const float* __restrict__ b_hm,
    const float* __restrict__ W_ih, const float* __restrict__ b_ih,
    const float* __restrict__ W_hh, const float* __restrict__ b_hh,
    _Float16* __restrict__ pred16)     // [B, T, 512]
{
  const int b = blockIdx.x;
  const int j = threadIdx.x;  // 0..255

  __shared__ float sh_h[HIDD];
  __shared__ float sh_ctx[HIDD];
  __shared__ float sh_x[HIDD];
  __shared__ float sh_hnew[HIDD];
  __shared__ float sh_gi[3 * HIDD];
  __shared__ float sh_gh[3 * HIDD];
  __shared__ float sh_sc[SRCL];
  __shared__ float sh_red[64];

  // stage initial hidden into sh_x, compute h0 = W_hm @ init + b_hm
  sh_x[j]   = h_init[(size_t)b * HIDD + j];
  sh_ctx[j] = 0.0f;
  __syncthreads();
  {
    const float* wr = W_hm + (size_t)j * HIDD;
    float acc = b_hm[j];
#pragma unroll 4
    for (int k = 0; k < HIDD; k += 4) {
      float4 w = *(const float4*)(wr + k);
      acc += w.x * sh_x[k] + w.y * sh_x[k + 1] + w.z * sh_x[k + 2] + w.w * sh_x[k + 3];
    }
    sh_h[j] = acc;
  }
  __syncthreads();

  const float* encb = enc + (size_t)b * SRCL * HIDD;

  for (int t = 0; t < TT; ++t) {
    // --- embedding (padding_idx = 0) ---
    int tok = tgt[(size_t)b * TT + t];
    sh_x[j] = (tok == 0) ? 0.0f : emb[(size_t)tok * EMBD + j];
    __syncthreads();

    // --- gi = W_ih @ [x; ctx] + b_ih ; gh = W_hh @ h + b_hh ---
    {
      const int KI = EMBD + HIDD;  // 512
      float a0 = b_ih[j], a1 = b_ih[j + HIDD], a2 = b_ih[j + 2 * HIDD];
      const float* w0 = W_ih + (size_t)j * KI;
      const float* w1 = W_ih + (size_t)(j + HIDD) * KI;
      const float* w2 = W_ih + (size_t)(j + 2 * HIDD) * KI;
#pragma unroll 4
      for (int k = 0; k < EMBD; k += 4) {
        float x0 = sh_x[k], x1 = sh_x[k + 1], x2 = sh_x[k + 2], x3 = sh_x[k + 3];
        float4 u = *(const float4*)(w0 + k); a0 += u.x * x0 + u.y * x1 + u.z * x2 + u.w * x3;
        float4 v = *(const float4*)(w1 + k); a1 += v.x * x0 + v.y * x1 + v.z * x2 + v.w * x3;
        float4 w = *(const float4*)(w2 + k); a2 += w.x * x0 + w.y * x1 + w.z * x2 + w.w * x3;
      }
#pragma unroll 4
      for (int k = 0; k < HIDD; k += 4) {
        float x0 = sh_ctx[k], x1 = sh_ctx[k + 1], x2 = sh_ctx[k + 2], x3 = sh_ctx[k + 3];
        float4 u = *(const float4*)(w0 + EMBD + k); a0 += u.x * x0 + u.y * x1 + u.z * x2 + u.w * x3;
        float4 v = *(const float4*)(w1 + EMBD + k); a1 += v.x * x0 + v.y * x1 + v.z * x2 + v.w * x3;
        float4 w = *(const float4*)(w2 + EMBD + k); a2 += w.x * x0 + w.y * x1 + w.z * x2 + w.w * x3;
      }
      sh_gi[j] = a0; sh_gi[HIDD + j] = a1; sh_gi[2 * HIDD + j] = a2;

      float g0 = b_hh[j], g1 = b_hh[j + HIDD], g2 = b_hh[j + 2 * HIDD];
      const float* v0 = W_hh + (size_t)j * HIDD;
      const float* v1 = W_hh + (size_t)(j + HIDD) * HIDD;
      const float* v2 = W_hh + (size_t)(j + 2 * HIDD) * HIDD;
#pragma unroll 4
      for (int k = 0; k < HIDD; k += 4) {
        float x0 = sh_h[k], x1 = sh_h[k + 1], x2 = sh_h[k + 2], x3 = sh_h[k + 3];
        float4 u = *(const float4*)(v0 + k); g0 += u.x * x0 + u.y * x1 + u.z * x2 + u.w * x3;
        float4 v = *(const float4*)(v1 + k); g1 += v.x * x0 + v.y * x1 + v.z * x2 + v.w * x3;
        float4 w = *(const float4*)(v2 + k); g2 += w.x * x0 + w.y * x1 + w.z * x2 + w.w * x3;
      }
      sh_gh[j] = g0; sh_gh[HIDD + j] = g1; sh_gh[2 * HIDD + j] = g2;
    }
    __syncthreads();

    // --- GRU gates ---
    float r  = sigmoidf_(sh_gi[j] + sh_gh[j]);
    float z  = sigmoidf_(sh_gi[HIDD + j] + sh_gh[HIDD + j]);
    float nn = tanhf(sh_gi[2 * HIDD + j] + r * sh_gh[2 * HIDD + j]);
    float hn = (1.0f - z) * nn + z * sh_h[j];
    sh_hnew[j] = hn;
    __syncthreads();

    // --- attention scores (threads 0..127) ---
    if (j < SRCL) {
      const float* er = encb + (size_t)j * HIDD;
      float acc = 0.0f;
#pragma unroll 4
      for (int k = 0; k < HIDD; k += 4) {
        float4 e = *(const float4*)(er + k);
        acc += e.x * sh_hnew[k] + e.y * sh_hnew[k + 1] + e.z * sh_hnew[k + 2] + e.w * sh_hnew[k + 3];
      }
      sh_sc[j] = acc;
    }
    __syncthreads();

    // --- softmax over 128 ---
    if (j < 64) sh_red[j] = fmaxf(sh_sc[j], sh_sc[j + 64]);
    __syncthreads();
    for (int off = 32; off > 0; off >>= 1) {
      if (j < off) sh_red[j] = fmaxf(sh_red[j], sh_red[j + off]);
      __syncthreads();
    }
    float mx = sh_red[0];
    if (j < SRCL) sh_sc[j] = __expf(sh_sc[j] - mx);
    __syncthreads();
    if (j < 64) sh_red[j] = sh_sc[j] + sh_sc[j + 64];
    __syncthreads();
    for (int off = 32; off > 0; off >>= 1) {
      if (j < off) sh_red[j] += sh_red[j + off];
      __syncthreads();
    }
    float inv = 1.0f / sh_red[0];

    // --- ctx = attn @ enc  (thread j owns hidden index j; coalesced over j) ---
    float cacc = 0.0f;
    for (int s = 0; s < SRCL; ++s) cacc += sh_sc[s] * encb[(size_t)s * HIDD + j];
    float cv = cacc * inv;
    sh_ctx[j] = cv;
    sh_h[j]   = hn;

    // --- emit pred = [ctx, h] as f16 at GEMM row m = b*T + t ---
    _Float16* pr = pred16 + ((size_t)b * TT + t) * KCLS;
    pr[j]        = (_Float16)cv;
    pr[HIDD + j] = (_Float16)hn;
    // next-iteration barrier at loop top covers sh_ctx / sh_h updates
  }
}

// ---------------------------------------------------------------------------
// Phase B: logits[2048, 32000] = pred16 @ Wc16^T + b_cls via v_wmma_f32_16x16x32_f16
// Each wave: one 32x64 output tile (2 M-tiles x 4 N-tiles -> 8 WMMAs / K-chunk,
// 12 b128 loads / K-chunk = 1.5 loads per WMMA; each B frag feeds 2 WMMAs).
// grid = 8000 blocks x 128 threads (4 waves); 64 M32-tiles * 500 N64-tiles.
// ---------------------------------------------------------------------------
__global__ __launch_bounds__(128) void classifier_gemm(
    const _Float16* __restrict__ A,   // pred16 [2048, 512], row-major
    const _Float16* __restrict__ Bw,  // Wc16   [32000, 512], row-major (N-major, K contiguous)
    const float*    __restrict__ bias,// [32000]
    float*          __restrict__ out) // [2048, 32000] == [B, T, V]
{
  const int wave = threadIdx.x >> 5;
  const int lane = threadIdx.x & 31;
  const int task = blockIdx.x * 4 + wave;  // 64 M32-tiles * 500 N64-tiles = 32000
  const int mt = task / 500;
  const int nt = task % 500;
  const int m0 = mt * 32;
  const int n0 = nt * 64;
  const int hpart = lane >> 4;   // 0: K-low half rows, 1: K-high half rows
  const int l     = lane & 15;

  v8f acc[4][2] = {};

  const _Float16* arow0 = A + (size_t)(m0 + l) * KCLS;
  const _Float16* arow1 = A + (size_t)(m0 + 16 + l) * KCLS;
  const _Float16* bc[4];
#pragma unroll
  for (int sj = 0; sj < 4; ++sj)
    bc[sj] = Bw + (size_t)(n0 + sj * 16 + l) * KCLS + hpart * 16;

  for (int k0 = 0; k0 < KCLS; k0 += 32) {
    if (k0 + 32 < KCLS) {
      __builtin_prefetch(bc[0] + k0 + 32, 0, 3);  // near-cache locality
      __builtin_prefetch(bc[2] + k0 + 32, 0, 3);
    }

    // A fragments: lane holds row m; contiguous K = k0 + hpart*16 .. +15
    v8h a0lo = *(const v8h*)(arow0 + k0 + hpart * 16);
    v8h a0hi = *(const v8h*)(arow0 + k0 + 8 + hpart * 16);
    v8h a1lo = *(const v8h*)(arow1 + k0 + hpart * 16);
    v8h a1hi = *(const v8h*)(arow1 + k0 + 8 + hpart * 16);
    v16h af0 = __builtin_shufflevector(a0lo, a0hi, 0, 1, 2, 3, 4, 5, 6, 7,
                                       8, 9, 10, 11, 12, 13, 14, 15);
    v16h af1 = __builtin_shufflevector(a1lo, a1hi, 0, 1, 2, 3, 4, 5, 6, 7,
                                       8, 9, 10, 11, 12, 13, 14, 15);

#pragma unroll
    for (int sj = 0; sj < 4; ++sj) {
      // B fragment: lane holds column n; contiguous K = k0 + hpart*16 .. +15
      v16h bf = *(const v16h*)(bc[sj] + k0);
      acc[sj][0] = __builtin_amdgcn_wmma_f32_16x16x32_f16(false, af0, false, bf,
                                                          (short)0, acc[sj][0], false, false);
      acc[sj][1] = __builtin_amdgcn_wmma_f32_16x16x32_f16(false, af1, false, bf,
                                                          (short)0, acc[sj][1], false, false);
    }
  }

  // Epilogue: C/D layout — VGPR r: lanes0-15 (M=r), lanes16-31 (M=8+r); N = lane&15.
#pragma unroll
  for (int sj = 0; sj < 4; ++sj) {
    const int col = n0 + sj * 16 + l;
    const float bv = bias[col];
#pragma unroll
    for (int half2 = 0; half2 < 2; ++half2) {
      const int rowbase = m0 + half2 * 16 + hpart * 8;
#pragma unroll
      for (int r = 0; r < 8; ++r) {
        __builtin_nontemporal_store(acc[sj][half2][r] + bv,
                                    &out[(size_t)(rowbase + r) * VOCAB + col]);
      }
    }
  }
}

// ---------------------------------------------------------------------------
extern "C" void kernel_launch(void* const* d_in, const int* in_sizes, int n_in,
                              void* d_out, int out_size, void* d_ws, size_t ws_size,
                              hipStream_t stream) {
  const float* enc   = (const float*)d_in[0];   // [B, SRC, H]
  const float* hinit = (const float*)d_in[1];   // [B, H]
  const int*   tgt   = (const int*)  d_in[2];   // [B, T]
  const float* emb   = (const float*)d_in[3];   // [VOCAB, E]
  const float* W_hm  = (const float*)d_in[4];
  const float* b_hm  = (const float*)d_in[5];
  const float* W_ih  = (const float*)d_in[6];
  const float* b_ih  = (const float*)d_in[7];
  const float* W_hh  = (const float*)d_in[8];
  const float* b_hh  = (const float*)d_in[9];
  const float* W_cls = (const float*)d_in[10];  // [V, 512]
  const float* b_cls = (const float*)d_in[11];  // [V]
  float* out = (float*)d_out;                   // [B, T, V] fp32

  // workspace layout
  char* ws = (char*)d_ws;
  _Float16* Wc16   = (_Float16*)ws;                       // 32000*512*2 = 32,768,000 B
  _Float16* pred16 = (_Float16*)(ws + 32768000);          // 2048*512*2  =  2,097,152 B

  // 0) W_cls -> f16 (resident in 192MB L2 thereafter)
  const int ncls = VOCAB * KCLS;  // 16,384,000 (divisible by 1024)
  cvt_wcls_f16<<<ncls / 1024, 256, 0, stream>>>(W_cls, Wc16, ncls);

  // 1) recurrent chain, one WG per batch element
  decoder_recurrent<<<BB, 256, 0, stream>>>(enc, hinit, tgt, emb,
                                            W_hm, b_hm, W_ih, b_ih, W_hh, b_hh,
                                            pred16);

  // 2) big classifier GEMM: 64 M32-tiles * 500 N64-tiles = 32000 wave-tasks, 4 waves/block
  classifier_gemm<<<32000 / 4, 128, 0, stream>>>(pred16, Wc16, b_cls, out);
}
